// OTNAE_6012954214797
// MI455X (gfx1250) — matrix operand order repaired and checked by reading, not compiled, not run on hardware
//
#include <hip/hip_runtime.h>

#define B_   2048
#define D_   128
#define H_   512
#define MT   32           // batch rows per workgroup
#define LDSW 260          // dwords per hid row: 512 halves + 8 pad halves
#define ENCW 130          // padded encT row stride (dwords)

typedef __attribute__((ext_vector_type(16))) __bf16       v16bf;
typedef __attribute__((ext_vector_type(2)))  __bf16       v2bf;
typedef __attribute__((ext_vector_type(8)))  float        v8f;
typedef __attribute__((ext_vector_type(4)))  unsigned int uint4v;
typedef __attribute__((ext_vector_type(2)))  unsigned int uint2v;
typedef __attribute__((ext_vector_type(4)))  float        float4v;
typedef __attribute__((ext_vector_type(2)))  float        float2v;

union ABFrag {
    struct { uint4v lo, hi; } u;   // 32 bytes = 16 bf16
    v16bf v;
};

union BFPack { v2bf h; unsigned int u; };

__device__ __forceinline__ unsigned int pack_bf16(float v0, float v1) {
    BFPack p;
    p.h = (v2bf){(__bf16)v0, (__bf16)v1};   // RNE; folds to v_cvt_pk_bf16_f32
    return p.u;
}

// branch-free tanh from HW transcendentals: v_exp_f32 + v_rcp_f32
__device__ __forceinline__ float fast_tanh(float x) {
    float a = __builtin_fabsf(x);
    // e^(-2a) = 2^(-2a*log2(e))
    float e = __builtin_amdgcn_exp2f(a * -2.885390081777927f);
    float r = (1.0f - e) * __builtin_amdgcn_rcpf(1.0f + e);
    return __builtin_copysignf(r, x);
}

// ---------------------------------------------------------------------------
// Kernel 1: encoder  encoded = tanh(leaky(x@We1^T + be1) @ We2^T + be2)
// one block of 256 threads handles 2 batch rows
// ---------------------------------------------------------------------------
__global__ __launch_bounds__(256) void otnae_encoder(
    const float* __restrict__ x,   const float* __restrict__ We1,
    const float* __restrict__ be1, const float* __restrict__ We2,
    const float* __restrict__ be2, float* __restrict__ enc)
{
    __shared__ float xsh[2][128];
    __shared__ float eh[2][512];
    const int t  = threadIdx.x;
    const int b0 = blockIdx.x * 2;

    xsh[t >> 7][t & 127] = x[(size_t)b0 * D_ + t];
    __syncthreads();

    #pragma unroll
    for (int k = 0; k < 4; ++k) {
        int task = t + (k << 8);
        int r = task >> 9;
        int h = task & 511;
        const float* wr = We1 + (size_t)h * D_;
        float s = be1[h];
        #pragma unroll 8
        for (int d = 0; d < 128; ++d) s = fmaf(xsh[r][d], wr[d], s);
        eh[r][h] = fmaxf(s, 0.2f * s);      // leaky relu
    }
    __syncthreads();

    {
        int r = t >> 7;
        int dcol = t & 127;
        const float* wr = We2 + (size_t)dcol * H_;
        float s = be2[dcol];
        #pragma unroll 8
        for (int h = 0; h < 512; ++h) s = fmaf(eh[r][h], wr[h], s);
        enc[(size_t)(b0 + r) * D_ + dcol] = fast_tanh(s);
    }
}

// ---------------------------------------------------------------------------
// Kernel 2: convert W2s (D,D,H) fp32 -> bf16 workspace copy
// ---------------------------------------------------------------------------
__global__ __launch_bounds__(256) void otnae_cvt_bf16(
    const float* __restrict__ src, unsigned int* __restrict__ dst, int n4)
{
    int i = blockIdx.x * 256 + threadIdx.x;
    if (i < n4) {
        float4v v = ((const float4v*)src)[i];
        uint2v p;
        p.x = pack_bf16(v.x, v.y);
        p.y = pack_bf16(v.z, v.w);
        ((uint2v*)dst)[i] = p;
    }
}

// ---------------------------------------------------------------------------
// Kernel 3: main fused decoder + cumsum + tanh
// grid = B/MT = 64 WGs; 256 threads = 8 waves, wave w owns d-columns [16w,16w+16)
// running WMMA f32 accumulators realize the inclusive prefix-sum over j
// ---------------------------------------------------------------------------
__global__ __launch_bounds__(256) void otnae_main(
    const float* __restrict__ enc,            // (B,D) fp32
    const float* __restrict__ W1s,            // (D,H) fp32
    const float* __restrict__ b1s,            // (D,H) fp32
    const unsigned short* __restrict__ W2bf,  // (D,D,H) bf16
    const float* __restrict__ b2s,            // (D,D) fp32
    float* __restrict__ out)                  // (D,B,D) fp32
{
    __shared__ unsigned int hid[MT * LDSW];   // 33,280 B  bf16 hid[32][512]+pad
    __shared__ float encT[MT * ENCW];         // 16,640 B  encoded rows of this tile

    const int t     = threadIdx.x;
    const int wave  = t >> 5;                 // 0..7 -> d-tile
    const int lane  = t & 31;
    const int lmod  = lane & 15;
    const int lhi   = lane >> 4;
    const int btile = blockIdx.x * MT;
    const int dcol  = (wave << 4) + lmod;

    // stage encoded tile (z values for all 128 j)
    for (int i = t; i < MT * D_; i += 256) {
        int r = i >> 7, c = i & 127;
        encT[r * ENCW + c] = enc[(size_t)(btile + r) * D_ + c];
    }

    const int row = t >> 3;                   // hid row this thread fills
    const int tr  = t & 7;

    v8f acc0a = {}, acc0b = {}, acc1a = {}, acc1b = {};
    float bsum = 0.f;

    const unsigned int* arow0 = hid + lmod * LDSW;
    const unsigned int* arow1 = hid + (lmod + 16) * LDSW;

    for (int j = 0; j < D_; ++j) {
        __syncthreads();   // prior-iteration A reads done before overwrite
        // ---- hid = leaky(z * W1s[j] + b1s[j]) -> LDS bf16 ----
        {
            const float z = encT[row * ENCW + j];
            const float2v* w2 = (const float2v*)(W1s + (size_t)j * H_);
            const float2v* b2 = (const float2v*)(b1s + (size_t)j * H_);
            unsigned int* hrow = hid + row * LDSW;
            #pragma unroll 8
            for (int i = 0; i < 32; ++i) {
                int dw = tr + (i << 3);       // dword index within row
                float2v wv = w2[dw];
                float2v bv = b2[dw];
                float v0 = fmaf(z, wv.x, bv.x);
                float v1 = fmaf(z, wv.y, bv.y);
                v0 = fmaxf(v0, 0.2f * v0);
                v1 = fmaxf(v1, 0.2f * v1);
                hrow[dw] = pack_bf16(v0, v1);
            }
        }
        __syncthreads();

        // ---- 16 K-tiles of v_wmma_f32_16x16x32_bf16, cumsum rides in C ----
        const unsigned short* Bj = W2bf + ((size_t)j * D_ + dcol) * H_;
        const float b2j = b2s[j * D_ + dcol];           // hoisted: hides under WMMAs
        __builtin_prefetch(Bj + (size_t)D_ * H_, 0, 1); // next-j slice (stays in ws)

        #pragma unroll 4
        for (int kt = 0; kt < 16; ++kt) {
            const int kb = kt << 5;                       // K base (h index)
            ABFrag bf, a0, a1;
            const uint4v* bp = (const uint4v*)(Bj + kb + (lhi << 4));
            bf.u.lo = bp[0];
            bf.u.hi = bp[1];
            const int kd = (kb >> 1) + (lhi << 3);        // dword offset, lo chunk
            a0.u.lo = *(const uint4v*)(arow0 + kd);
            a0.u.hi = *(const uint4v*)(arow0 + kd + 4);
            a1.u.lo = *(const uint4v*)(arow1 + kd);
            a1.u.hi = *(const uint4v*)(arow1 + kd + 4);
            if (kt & 1) {
                acc0b = __builtin_amdgcn_wmma_f32_16x16x32_bf16(
                    false, a0.v, false, bf.v, (short)0, acc0b, false, false);
                acc1b = __builtin_amdgcn_wmma_f32_16x16x32_bf16(
                    false, a1.v, false, bf.v, (short)0, acc1b, false, false);
            } else {
                acc0a = __builtin_amdgcn_wmma_f32_16x16x32_bf16(
                    false, a0.v, false, bf.v, (short)0, acc0a, false, false);
                acc1a = __builtin_amdgcn_wmma_f32_16x16x32_bf16(
                    false, a1.v, false, bf.v, (short)0, acc1a, false, false);
            }
        }

        bsum += b2j;                                      // bias part of cumsum

        // ---- emit outputs[j] = tanh(running sum), branch-free tanh ----
        float* oj = out + (size_t)j * B_ * D_ + (size_t)btile * D_ + dcol;
        #pragma unroll
        for (int r = 0; r < 8; ++r) {
            int m0 = r + (lhi << 3);                      // C layout: M = r + 8*(lane/16)
            oj[(size_t)m0 * D_]        = fast_tanh(acc0a[r] + acc0b[r] + bsum);
            oj[(size_t)(m0 + 16) * D_] = fast_tanh(acc1a[r] + acc1b[r] + bsum);
        }
    }
}

// ---------------------------------------------------------------------------
extern "C" void kernel_launch(void* const* d_in, const int* in_sizes, int n_in,
                              void* d_out, int out_size, void* d_ws, size_t ws_size,
                              hipStream_t stream)
{
    const float* x   = (const float*)d_in[0];
    const float* We1 = (const float*)d_in[1];
    const float* be1 = (const float*)d_in[2];
    const float* We2 = (const float*)d_in[3];
    const float* be2 = (const float*)d_in[4];
    const float* W1s = (const float*)d_in[5];
    const float* b1s = (const float*)d_in[6];
    const float* W2s = (const float*)d_in[7];
    const float* b2s = (const float*)d_in[8];
    float* out = (float*)d_out;

    const size_t w2_elems = (size_t)D_ * D_ * H_;             // 8,388,608
    unsigned short* W2bf = (unsigned short*)d_ws;             // 16 MiB bf16
    float* enc = (float*)((char*)d_ws + w2_elems * sizeof(unsigned short));

    otnae_encoder<<<B_ / 2, 256, 0, stream>>>(x, We1, be1, We2, be2, enc);

    int n4 = (int)(w2_elems / 4);                             // 2,097,152
    otnae_cvt_bf16<<<(n4 + 255) / 256, 256, 0, stream>>>(W2s, (unsigned int*)W2bf, n4);

    otnae_main<<<B_ / MT, 256, 0, stream>>>(enc, W1s, b1s, W2bf, b2s, out);
}
